// GraphHyperbolicVisitEncoderGlobal_20031727469141
// MI455X (gfx1250) — matrix operand order repaired and checked by reading, not compiled, not run on hardware
//
#include <hip/hip_runtime.h>
#include <math.h>
#include <stdint.h>

// Problem constants (from reference)
#define V_   4096
#define D_   128
#define K_   3
#define NH_  4
#define FF_  512
#define NV_  16384
#define MC_  32
#define NL_  2
#define HD_  32   // D_/NH_

typedef __attribute__((ext_vector_type(16))) __bf16 v16bf;
typedef __attribute__((ext_vector_type(8)))  __bf16 v8bf;
typedef __attribute__((ext_vector_type(8)))  float  v8f;
typedef __attribute__((ext_vector_type(4)))  unsigned int u32x4;
typedef __attribute__((ext_vector_type(8)))  int i32x8;
typedef __attribute__((ext_vector_type(4)))  int i32x4;

// fp32 -> bf16 via native conversion (v_cvt_pk_bf16_f32 on gfx1250)
__device__ __forceinline__ __bf16 f2bf(float f) { return (__bf16)f; }

// ---- WMMA fragment loaders (wave32, 16x16x32 bf16) -------------------------
// A fragment: 16x32 (MxK). Lane L: m = L&15, half = L>>4.
//   element j -> K = j + (j>=8 ? 8 : 0) + half*8   (two contiguous runs of 8)
__device__ __forceinline__ v16bf load_a_f32(const float* A, int lda, int row0,
                                            int k0, int lane) {
  const int m  = lane & 15;
  const int h8 = (lane >> 4) << 3;
  const float* p = A + (size_t)(row0 + m) * (size_t)lda + (size_t)(k0 + h8);
  float t[16];
  *(float4*)(t + 0)  = *(const float4*)(p + 0);
  *(float4*)(t + 4)  = *(const float4*)(p + 4);
  *(float4*)(t + 8)  = *(const float4*)(p + 16);
  *(float4*)(t + 12) = *(const float4*)(p + 20);
  v16bf a;
#pragma unroll
  for (int j = 0; j < 16; ++j) a[j] = f2bf(t[j]);
  return a;
}

__device__ __forceinline__ v16bf load_a_bf16(const __bf16* A, int lda, int row0,
                                             int k0, int lane) {
  const int m  = lane & 15;
  const int h8 = (lane >> 4) << 3;
  const __bf16* p = A + (size_t)(row0 + m) * (size_t)lda + (size_t)(k0 + h8);
  v8bf lo = *(const v8bf*)(p);
  v8bf hi = *(const v8bf*)(p + 16);
  v16bf a;
#pragma unroll
  for (int j = 0; j < 8; ++j) { a[j] = lo[j]; a[8 + j] = hi[j]; }
  return a;
}

// B fragment: 32x16 (KxN) from K-contiguous buffer BT[N][Kd].
// Lane L: n = L&15, element j -> K = j + (j>=8 ? 8 : 0) + (L>>4)*8 (two runs of 8).
__device__ __forceinline__ v16bf load_b_bf16(const __bf16* BT, int ldb, int col0,
                                             int k0, int lane) {
  const int n  = lane & 15;
  const int kh = (lane >> 4) << 3;
  const __bf16* p = BT + (size_t)(col0 + n) * (size_t)ldb + (size_t)(k0 + kh);
  v8bf lo = *(const v8bf*)(p);
  v8bf hi = *(const v8bf*)(p + 16);
  v16bf b;
#pragma unroll
  for (int j = 0; j < 8; ++j) { b[j] = lo[j]; b[8 + j] = hi[j]; }
  return b;
}

__device__ __forceinline__ v8f wmma_bf16(v16bf a, v16bf b, v8f c) {
  return __builtin_amdgcn_wmma_f32_16x16x32_bf16(false, a, false, b,
                                                 (short)0, c, false, false);
}

__device__ __forceinline__ int rfl(unsigned v) {
  return __builtin_amdgcn_readfirstlane((int)v);
}

// Issue one TDM 2D-tile transfer: 16 rows x KCbytes of bf16, global -> LDS.
// Descriptor per CDNA5 ISA (group0 + group1, 2D tile).
__device__ __forceinline__ void tdm_issue_16rows(const __bf16* gptr,
                                                 unsigned lds_byte,
                                                 int tile_k /*elements*/) {
  uint64_t gaddr = (uint64_t)(uintptr_t)gptr;
  u32x4 g0;
  g0[0] = (unsigned)rfl(1u);                               // count=1, user mode
  g0[1] = (unsigned)rfl(lds_byte);                         // lds_addr (bytes)
  g0[2] = (unsigned)rfl((unsigned)(gaddr & 0xFFFFFFFFu));  // global_addr lo
  g0[3] = (unsigned)rfl((unsigned)((gaddr >> 32) & 0x1FFFFFFu) | (2u << 30)); // hi|type=2
  i32x8 g1;
  g1[0] = rfl(1u << 16);                                   // data_size=1 (2 bytes)
  g1[1] = rfl(((unsigned)V_ & 0xFFFFu) << 16);             // tensor_dim0 lo16
  g1[2] = rfl(((unsigned)V_ >> 16) | (((unsigned)D_ & 0xFFFFu) << 16)); // d0 hi|d1 lo
  g1[3] = rfl(((unsigned)D_ >> 16) | ((unsigned)tile_k << 16));         // d1 hi|tile_dim0
  g1[4] = rfl(16u);                                        // tile_dim1=16 rows
  g1[5] = rfl((unsigned)V_);                               // tensor_dim0_stride lo32
  g1[6] = rfl(0u);
  g1[7] = rfl(0u);
  i32x4 g2; g2[0] = 0; g2[1] = 0; g2[2] = 0; g2[3] = 0;
  i32x4 g3; g3[0] = 0; g3[1] = 0; g3[2] = 0; g3[3] = 0;
#if defined(__clang_major__) && (__clang_major__ >= 23)
  i32x8 g4;
#pragma unroll
  for (int j = 0; j < 8; ++j) g4[j] = 0;
  __builtin_amdgcn_tensor_load_to_lds(g0, g1, g2, g3, g4, 0);
#else
  __builtin_amdgcn_tensor_load_to_lds(g0, g1, g2, g3, 0);
#endif
}

// ---- generic WMMA GEMM (software-pipelined K loop) -------------------------
// C[M][N] = A[M][Kd] * BT[N][Kd]^T (+bias)(+gelu). 4 waves, 16x64 per wave.
template <bool AF32>
__global__ void wmma_gemm_kernel(const void* __restrict__ Aptr,
                                 const __bf16* __restrict__ BT,
                                 int M, int N, int Kd, int lda,
                                 const float* __restrict__ bias, int act,
                                 float* __restrict__ Cf, int ldc,
                                 __bf16* __restrict__ Cb, int ldcb, int cb_trans) {
  const int lane = threadIdx.x & 31;
  const int wave = threadIdx.x >> 5;
  const int row0 = blockIdx.y * 64 + wave * 16;
  const int col0 = blockIdx.x * 64;
  const float*  Af = (const float*)Aptr;
  const __bf16* Ab = (const __bf16*)Aptr;

  v8f acc[4];
#pragma unroll
  for (int t = 0; t < 4; ++t)
#pragma unroll
    for (int r = 0; r < 8; ++r) acc[t][r] = 0.0f;

  auto loadA = [&](int k0) -> v16bf {
    if constexpr (AF32) return load_a_f32(Af, lda, row0, k0, lane);
    else                return load_a_bf16(Ab, lda, row0, k0, lane);
  };

  v16bf a0 = loadA(0);
  v16bf b0[4], b1[4];
#pragma unroll
  for (int t = 0; t < 4; ++t) b0[t] = load_b_bf16(BT, Kd, col0 + 16 * t, 0, lane);

  for (int k0 = 0; k0 + 32 < Kd; k0 += 32) {
    if constexpr (AF32) {
      __builtin_prefetch(
        (const void*)(Af + (size_t)(row0 + (lane & 15)) * (size_t)lda + k0 + 160),
        0, 1);
    }
    v16bf a1 = loadA(k0 + 32);
#pragma unroll
    for (int t = 0; t < 4; ++t)
      b1[t] = load_b_bf16(BT, Kd, col0 + 16 * t, k0 + 32, lane);
#pragma unroll
    for (int t = 0; t < 4; ++t) acc[t] = wmma_bf16(a0, b0[t], acc[t]);
    a0 = a1;
#pragma unroll
    for (int t = 0; t < 4; ++t) b0[t] = b1[t];
  }
#pragma unroll
  for (int t = 0; t < 4; ++t) acc[t] = wmma_bf16(a0, b0[t], acc[t]);

  const int nlo  = lane & 15;
  const int half = lane >> 4;
#pragma unroll
  for (int t = 0; t < 4; ++t) {
#pragma unroll
    for (int r = 0; r < 8; ++r) {
      int row = row0 + r + half * 8;
      int col = col0 + t * 16 + nlo;
      float v = acc[t][r];
      if (bias) v += bias[col];
      if (act == 1) v = 0.5f * v * (1.0f + erff(v * 0.70710678118654752f));
      if (Cf) Cf[(size_t)row * (size_t)ldc + col] = v;
      if (Cb) {
        __bf16 bv = f2bf(v);
        if (cb_trans) Cb[(size_t)col * (size_t)ldcb + row] = bv;
        else          Cb[(size_t)row * (size_t)ldcb + col] = bv;
      }
    }
  }
}

// ---- diffusion GEMM: TDM double-buffered B panel, full-N block -------------
// Zk_part[V][128] = kernels_k[V][V] (fp32, streamed once) @ Z0 (Z0T bf16).
// 256 threads = 8 waves (4 row-groups x 2 col-groups); block tile 64M x 128N,
// so each A row is fetched by one block only (~201 MB total HBM stream).
// Each wave TDM-loads 16 of the 128 B columns per K-chunk; chunk i+1's DMA
// overlaps chunk i's WMMA via double buffering + s_wait_tensorcnt(1).
__global__ __launch_bounds__(256)
void wmma_gemm_diffusion_tdm(const float* __restrict__ A,
                             const __bf16* __restrict__ BT,   // Z0T [128][V]
                             __bf16* __restrict__ Cb, int ldcb) {
  constexpr int KC  = 128;        // K elements staged per chunk
  constexpr int NCH = V_ / KC;    // 32 chunks
  __shared__ __align__(32) __bf16 ldsB[2][D_ * KC];  // 2 x 32 KB: [col][KC]
  const int lane = threadIdx.x & 31;
  const int wave = threadIdx.x >> 5;
  const int row0 = blockIdx.y * 64 + (wave & 3) * 16;
  const int col0 = (wave >> 2) * 64;

  // this wave's TDM job: B-panel columns [16*wave, 16*wave+16)
  const __bf16* gsrc = BT + (size_t)(wave * 16) * (size_t)V_;
  const unsigned ldsw0 =
      (unsigned)(uintptr_t)(&ldsB[0][0]) + (unsigned)(wave * 16 * KC * 2);
  const unsigned ldsw1 =
      (unsigned)(uintptr_t)(&ldsB[1][0]) + (unsigned)(wave * 16 * KC * 2);

  v8f acc[4];
#pragma unroll
  for (int t = 0; t < 4; ++t)
#pragma unroll
    for (int r = 0; r < 8; ++r) acc[t][r] = 0.0f;

  // prologue: start DMA of chunk 0
  tdm_issue_16rows(gsrc, ldsw0, KC);

  for (int i = 0; i < NCH; ++i) {
    const int kc = i * KC;
    if (i + 1 < NCH) {
      // kick off next chunk's DMA into the other buffer, then wait until only
      // that transfer is outstanding -> current buffer is complete
      tdm_issue_16rows(gsrc + kc + KC, ((i + 1) & 1) ? ldsw1 : ldsw0, KC);
      __builtin_amdgcn_s_wait_tensorcnt(1);
    } else {
      __builtin_amdgcn_s_wait_tensorcnt(0);
    }
    __syncthreads();  // all waves' sub-tiles of buf[i&1] are resident

    const __bf16* bufB = &ldsB[i & 1][0];
#pragma unroll
    for (int kk = 0; kk < KC; kk += 32) {
      __builtin_prefetch(
        (const void*)(A + (size_t)(row0 + (lane & 15)) * (size_t)V_ + kc + kk + 512),
        0, 1);
      v16bf a = load_a_f32(A, V_, row0, kc + kk, lane);
#pragma unroll
      for (int t = 0; t < 4; ++t) {
        v16bf b = load_b_bf16(bufB, KC, col0 + t * 16, kk, lane);
        acc[t] = wmma_bf16(a, b, acc[t]);
      }
    }
    __syncthreads();  // everyone done reading buf[i&1] before it is re-filled
  }

  const int nlo  = lane & 15;
  const int half = lane >> 4;
#pragma unroll
  for (int t = 0; t < 4; ++t) {
#pragma unroll
    for (int r = 0; r < 8; ++r) {
      int row = row0 + r + half * 8;
      int col = col0 + t * 16 + nlo;
      Cb[(size_t)row * (size_t)ldcb + col] = f2bf(acc[t][r]);
    }
  }
}

// ---- transpose + convert fp32[R][C] -> bf16[C][R] (small weight matrices) --
__global__ void transpose_f32_bf16_kernel(const float* __restrict__ src,
                                          __bf16* __restrict__ dst, int R, int C) {
  int idx = blockIdx.x * blockDim.x + threadIdx.x;
  if (idx < R * C) {
    int r = idx / C, c = idx % C;
    dst[(size_t)c * R + r] = f2bf(src[(size_t)r * C + c]);
  }
}

// ---- hyperbolic logmap0, writes Z0^T in bf16 [D][V] ------------------------
__global__ void logmap_kernel(const float* __restrict__ X, __bf16* __restrict__ Z0T) {
  const int v = blockIdx.x, d = threadIdx.x;
  __shared__ float red[D_];
  float x = X[(size_t)v * D_ + d];
  red[d] = x * x;
  __syncthreads();
  for (int s = D_ / 2; s > 0; s >>= 1) {
    if (d < s) red[d] += red[d + s];
    __syncthreads();
  }
  float n = sqrtf(red[0]);
  n = fminf(fmaxf(n, 1e-7f), 1.0f - 1e-5f);
  float scl = 0.5f * logf((1.0f + n) / (1.0f - n)) / n;  // atanh(n)/n
  Z0T[(size_t)d * V_ + v] = f2bf(scl * x);
}

// ---- flash attention: one wave per (16-query tile, head) -------------------
__global__ void flash_attn_kernel(const __bf16* __restrict__ q,
                                  const __bf16* __restrict__ k,
                                  const __bf16* __restrict__ vT,  // [D][V]
                                  __bf16* __restrict__ o) {
  const int lane = threadIdx.x;
  const int h    = blockIdx.y;
  const int q0   = blockIdx.x * 16;
  const int nlo  = lane & 15;
  const int half = lane >> 4;
  __shared__ __align__(32) __bf16 pls[16 * 32];

  v16bf qa = load_a_bf16(q + h * HD_, D_, q0, 0, lane);

  float Mrow[8], Lrow[8];
  v8f o0, o1;
#pragma unroll
  for (int r = 0; r < 8; ++r) {
    Mrow[r] = -3.0e38f; Lrow[r] = 0.0f; o0[r] = 0.0f; o1[r] = 0.0f;
  }
  const float sc = 0.17677669529663687f;  // 1/sqrt(HD)

  for (int kb = 0; kb < V_; kb += 32) {
    v8f z;
#pragma unroll
    for (int r = 0; r < 8; ++r) z[r] = 0.0f;
    v16bf b0 = load_b_bf16(k + h * HD_, D_, kb,      0, lane);
    v16bf b1 = load_b_bf16(k + h * HD_, D_, kb + 16, 0, lane);
    v8f s0 = wmma_bf16(qa, b0, z);
    v8f s1 = wmma_bf16(qa, b1, z);

#pragma unroll
    for (int r = 0; r < 8; ++r) {
      float a0 = s0[r] * sc, a1 = s1[r] * sc;
      float mx = fmaxf(a0, a1);
      mx = fmaxf(mx, __shfl_xor(mx, 1));
      mx = fmaxf(mx, __shfl_xor(mx, 2));
      mx = fmaxf(mx, __shfl_xor(mx, 4));
      mx = fmaxf(mx, __shfl_xor(mx, 8));
      float mn   = fmaxf(Mrow[r], mx);
      float corr = __expf(Mrow[r] - mn);
      float p0   = __expf(a0 - mn);
      float p1   = __expf(a1 - mn);
      float ls = p0 + p1;
      ls += __shfl_xor(ls, 1);
      ls += __shfl_xor(ls, 2);
      ls += __shfl_xor(ls, 4);
      ls += __shfl_xor(ls, 8);
      Lrow[r] = Lrow[r] * corr + ls;
      Mrow[r] = mn;
      o0[r] *= corr; o1[r] *= corr;
      int row = r + half * 8;
      pls[row * 32 + nlo]      = f2bf(p0);
      pls[row * 32 + 16 + nlo] = f2bf(p1);
    }
    __syncthreads();
    v16bf pa  = load_a_bf16(pls, 32, 0, 0, lane);
    v16bf vb0 = load_b_bf16(vT + (size_t)h * HD_ * V_, V_, 0,  kb, lane);
    v16bf vb1 = load_b_bf16(vT + (size_t)h * HD_ * V_, V_, 16, kb, lane);
    o0 = wmma_bf16(pa, vb0, o0);
    o1 = wmma_bf16(pa, vb1, o1);
    __syncthreads();
  }

#pragma unroll
  for (int r = 0; r < 8; ++r) {
    float inv = 1.0f / Lrow[r];
    int row = q0 + r + half * 8;
    o[(size_t)row * D_ + h * HD_ + nlo]      = f2bf(o0[r] * inv);
    o[(size_t)row * D_ + h * HD_ + 16 + nlo] = f2bf(o1[r] * inv);
  }
}

// ---- residual + LayerNorm; updates x (fp32) and its bf16 mirror ------------
__global__ void add_ln_kernel(float* __restrict__ x, const float* __restrict__ t,
                              const float* __restrict__ g, const float* __restrict__ b,
                              __bf16* __restrict__ xbf) {
  const int v = blockIdx.x, d = threadIdx.x;
  __shared__ float red[D_];
  __shared__ float stat;
  float val = x[(size_t)v * D_ + d] + t[(size_t)v * D_ + d];
  red[d] = val;
  __syncthreads();
  for (int s = D_ / 2; s > 0; s >>= 1) { if (d < s) red[d] += red[d + s]; __syncthreads(); }
  if (d == 0) stat = red[0] * (1.0f / D_);
  __syncthreads();
  float mean = stat;
  float cc = val - mean;
  red[d] = cc * cc;
  __syncthreads();
  for (int s = D_ / 2; s > 0; s >>= 1) { if (d < s) red[d] += red[d + s]; __syncthreads(); }
  if (d == 0) stat = red[0] * (1.0f / D_);
  __syncthreads();
  float y = cc * rsqrtf(stat + 1e-5f) * g[d] + b[d];
  x[(size_t)v * D_ + d] = y;
  xbf[(size_t)v * D_ + d] = f2bf(y);
}

// ---- time features: G = 1 - tanh(delta/180 * tf_W + tf_b)^2 (bf16) ---------
__global__ void time_feat_kernel(const float* __restrict__ deltas,
                                 const float* __restrict__ tf_W,
                                 const float* __restrict__ tf_b,
                                 __bf16* __restrict__ G) {
  const int nv = blockIdx.x, d = threadIdx.x;
  float f = deltas[nv] * (1.0f / 180.0f) * tf_W[d] + tf_b[d];
  float t = tanhf(f);
  G[(size_t)nv * D_ + d] = f2bf(1.0f - t * t);
}

// ---- ragged gather + masked mean + add time embedding ----------------------
__global__ void gather_mean_kernel(const int* __restrict__ visits,
                                   const int* __restrict__ pad_ptr,
                                   const float* __restrict__ Hr,
                                   const float* __restrict__ te,
                                   float* __restrict__ out) {
  const int nv = blockIdx.x, d = threadIdx.x;
  const int pad = pad_ptr[0];
  float acc = 0.0f;
  int cnt = 0;
#pragma unroll 4
  for (int c = 0; c < MC_; ++c) {
    int id = visits[(size_t)nv * MC_ + c];
    if (id != pad) { acc += Hr[(size_t)id * D_ + d]; ++cnt; }
  }
  out[(size_t)nv * D_ + d] = acc / (float)max(cnt, 1) + te[(size_t)nv * D_ + d];
}

// ---------------------------------------------------------------------------
extern "C" void kernel_launch(void* const* d_in, const int* in_sizes, int n_in,
                              void* d_out, int out_size, void* d_ws, size_t ws_size,
                              hipStream_t stream) {
  (void)in_sizes; (void)n_in; (void)out_size; (void)ws_size;
  const int*   visits  = (const int*)  d_in[0];
  const float* deltas  = (const float*)d_in[1];
  const int*   pad_idx = (const int*)  d_in[2];
  const float* X_hyp   = (const float*)d_in[3];
  const float* kernels = (const float*)d_in[4];
  const float* proj_W  = (const float*)d_in[5];
  const float* proj_b  = (const float*)d_in[6];
  const float* Wq      = (const float*)d_in[7];
  const float* Wk      = (const float*)d_in[8];
  const float* Wv      = (const float*)d_in[9];
  const float* Wo      = (const float*)d_in[10];
  const float* bq      = (const float*)d_in[11];
  const float* bk      = (const float*)d_in[12];
  const float* bv      = (const float*)d_in[13];
  const float* bo      = (const float*)d_in[14];
  const float* ff_W1   = (const float*)d_in[15];
  const float* ff_b1   = (const float*)d_in[16];
  const float* ff_W2   = (const float*)d_in[17];
  const float* ff_b2   = (const float*)d_in[18];
  const float* ln1_g   = (const float*)d_in[19];
  const float* ln1_b   = (const float*)d_in[20];
  const float* ln2_g   = (const float*)d_in[21];
  const float* ln2_b   = (const float*)d_in[22];
  const float* tf_W    = (const float*)d_in[23];
  const float* tf_b    = (const float*)d_in[24];
  const float* tp_W    = (const float*)d_in[25];
  const float* tp_b    = (const float*)d_in[26];
  float* out = (float*)d_out;

  // workspace bump allocator (256B aligned)
  char* w = (char*)d_ws;
  size_t off = 0;
  auto walloc = [&](size_t bytes) -> void* {
    void* p = w + off;
    off = (off + bytes + 255) & ~(size_t)255;
    return p;
  };
  __bf16* Z0T    = (__bf16*)walloc((size_t)D_ * V_ * 2);           // [D][V]
  __bf16* projWT = (__bf16*)walloc((size_t)D_ * K_ * D_ * 2);      // [128][384]
  __bf16* WT     = (__bf16*)walloc((size_t)NL_ * 4 * D_ * D_ * 2); // [l][q,k,v,o][out][in]
  __bf16* ff1T   = (__bf16*)walloc((size_t)NL_ * FF_ * D_ * 2);    // [l][512][128]
  __bf16* ff2T   = (__bf16*)walloc((size_t)NL_ * D_ * FF_ * 2);    // [l][128][512]
  __bf16* tpWT   = (__bf16*)walloc((size_t)D_ * D_ * 2);
  __bf16* Zk     = (__bf16*)walloc((size_t)V_ * K_ * D_ * 2);      // [V][384]
  float*  x      = (float*) walloc((size_t)V_ * D_ * 4);
  __bf16* xbf    = (__bf16*)walloc((size_t)V_ * D_ * 2);
  __bf16* qbf    = (__bf16*)walloc((size_t)V_ * D_ * 2);
  __bf16* kbf    = (__bf16*)walloc((size_t)V_ * D_ * 2);
  __bf16* vTbf   = (__bf16*)walloc((size_t)D_ * V_ * 2);           // [D][V]
  __bf16* obf    = (__bf16*)walloc((size_t)V_ * D_ * 2);
  float*  tbuf   = (float*) walloc((size_t)V_ * D_ * 4);
  __bf16* ffh    = (__bf16*)walloc((size_t)V_ * FF_ * 2);
  __bf16* G      = (__bf16*)walloc((size_t)NV_ * D_ * 2);
  float*  te     = (float*) walloc((size_t)NV_ * D_ * 4);

  auto transp = [&](const float* src, __bf16* dst, int R, int C) {
    int n = R * C;
    transpose_f32_bf16_kernel<<<(n + 255) / 256, 256, 0, stream>>>(src, dst, R, C);
  };
  auto gemm_bf = [&](const __bf16* Ab, const __bf16* Bt,
                     int M, int N, int Kd, int lda, const float* bias, int act,
                     float* Cf, int ldc, __bf16* Cb, int ldcb, int cb_trans) {
    dim3 g(N / 64, M / 64);
    wmma_gemm_kernel<false><<<g, 128, 0, stream>>>(
        (const void*)Ab, Bt, M, N, Kd, lda, bias, act, Cf, ldc, Cb, ldcb, cb_trans);
  };

  // --- weight prep: transpose+convert to bf16 K-contiguous layouts ---
  transp(proj_W, projWT, K_ * D_, D_);
  for (int l = 0; l < NL_; ++l) {
    transp(Wq + (size_t)l * D_ * D_, WT + ((size_t)l * 4 + 0) * D_ * D_, D_, D_);
    transp(Wk + (size_t)l * D_ * D_, WT + ((size_t)l * 4 + 1) * D_ * D_, D_, D_);
    transp(Wv + (size_t)l * D_ * D_, WT + ((size_t)l * 4 + 2) * D_ * D_, D_, D_);
    transp(Wo + (size_t)l * D_ * D_, WT + ((size_t)l * 4 + 3) * D_ * D_, D_, D_);
    transp(ff_W1 + (size_t)l * D_ * FF_, ff1T + (size_t)l * FF_ * D_, D_, FF_);
    transp(ff_W2 + (size_t)l * FF_ * D_, ff2T + (size_t)l * D_ * FF_, FF_, D_);
  }
  transp(tp_W, tpWT, D_, D_);

  // --- logmap0 -> Z0^T (bf16) ---
  logmap_kernel<<<V_, D_, 0, stream>>>(X_hyp, Z0T);

  // --- diffusion (streams 201 MB fp32 once; TDM double-buffered B) ---
  for (int kk = 0; kk < K_; ++kk) {
    wmma_gemm_diffusion_tdm<<<dim3(1, V_ / 64), 256, 0, stream>>>(
        kernels + (size_t)kk * V_ * V_, Z0T, Zk + kk * D_, K_ * D_);
  }

  // --- projection: x = Zk @ proj_W + proj_b ---
  gemm_bf(Zk, projWT, V_, D_, K_ * D_, K_ * D_, proj_b, 0, x, D_, xbf, D_, 0);

  // --- transformer layers ---
  for (int l = 0; l < NL_; ++l) {
    const __bf16* WqT = WT + ((size_t)l * 4 + 0) * D_ * D_;
    const __bf16* WkT = WT + ((size_t)l * 4 + 1) * D_ * D_;
    const __bf16* WvT = WT + ((size_t)l * 4 + 2) * D_ * D_;
    const __bf16* WoT = WT + ((size_t)l * 4 + 3) * D_ * D_;
    gemm_bf(xbf, WqT, V_, D_, D_, D_, bq + (size_t)l * D_, 0, nullptr, 0, qbf, D_, 0);
    gemm_bf(xbf, WkT, V_, D_, D_, D_, bk + (size_t)l * D_, 0, nullptr, 0, kbf, D_, 0);
    gemm_bf(xbf, WvT, V_, D_, D_, D_, bv + (size_t)l * D_, 0, nullptr, 0, vTbf, V_, 1);
    flash_attn_kernel<<<dim3(V_ / 16, NH_), 32, 0, stream>>>(qbf, kbf, vTbf, obf);
    gemm_bf(obf, WoT, V_, D_, D_, D_, bo + (size_t)l * D_, 0, tbuf, D_, nullptr, 0, 0);
    add_ln_kernel<<<V_, D_, 0, stream>>>(x, tbuf, ln1_g + (size_t)l * D_,
                                         ln1_b + (size_t)l * D_, xbf);
    gemm_bf(xbf, ff1T + (size_t)l * FF_ * D_, V_, FF_, D_, D_,
            ff_b1 + (size_t)l * FF_, 1 /*gelu*/, nullptr, 0, ffh, FF_, 0);
    gemm_bf(ffh, ff2T + (size_t)l * D_ * FF_, V_, D_, FF_, FF_,
            ff_b2 + (size_t)l * D_, 0, tbuf, D_, nullptr, 0, 0);
    add_ln_kernel<<<V_, D_, 0, stream>>>(x, tbuf, ln2_g + (size_t)l * D_,
                                         ln2_b + (size_t)l * D_, xbf);
  }

  // --- time embedding: te = (1 - tanh(f)^2) @ tp_W + tp_b ---
  time_feat_kernel<<<NV_, D_, 0, stream>>>(deltas, tf_W, tf_b, G);
  gemm_bf(G, tpWT, NV_, D_, D_, D_, tp_b, 0, te, D_, nullptr, 0, 0);

  // --- ragged gather + masked mean + te ---
  gather_mean_kernel<<<NV_, D_, 0, stream>>>(visits, pad_idx, x, te, out);
}
